// LSTM_DOUBLE_ATT_STACK_PARALLEL_57990648430773
// MI455X (gfx1250) — compile-verified, matrix operand order checked by reading, not run on hardware
//
#include <hip/hip_runtime.h>
#include <hip/hip_bf16.h>

// Problem constants
#define L_   2
#define P_   3
#define R_   512
#define A_   196
#define V_   9487
#define B_   256
#define BA_  (B_*A_)      // 50176 flattened (b,a) rows
#define G4_  (4*R_)       // 2048 gates per bank
#define NG_  (P_*G4_)     // 6144 gate columns per layer
#define AP_  208          // A padded to multiple of 16 (13 N-tiles = 3x4 + 1)
#define VP_  9536         // V padded to multiple of 64 -> 149 full groups of 4
#define NTV_ (VP_/16)     // 596 N-tiles in projection
#define VGRP_ (NTV_/4)    // 149 groups, all full
#define BR_  (B_*R_)      // 131072

typedef __attribute__((ext_vector_type(16))) _Float16 v16h;
typedef __attribute__((ext_vector_type(8)))  _Float16 v8h;
typedef __attribute__((ext_vector_type(4)))  _Float16 v4h;
typedef __attribute__((ext_vector_type(8)))  float    v8f;

// ---------------------------------------------------------------------------
// WMMA helpers (gfx1250 wave32, V_WMMA_F32_16X16X32_F16)
// A-frag (16x32 f16): lane&15 = M row; lanes<16 hold K=0..7 & 16..23,
// lanes>=16 hold K=8..15 & 24..31 (ISA 7.12.2). B read column-major, mirrored.
// ---------------------------------------------------------------------------
__device__ __forceinline__ v16h load_frag(const _Float16* __restrict__ base, int ld,
                                          int row0, int k0, int lane) {
  int r  = row0 + (lane & 15);
  int kb = (lane < 16) ? 0 : 8;
  const _Float16* p = base + (size_t)r * ld + (size_t)(k0 + kb);
  v8h lo = *(const v8h*)(p);
  v8h hi = *(const v8h*)(p + 16);
  return __builtin_shufflevector(lo, hi, 0,1,2,3,4,5,6,7,8,9,10,11,12,13,14,15);
}

__device__ __forceinline__ v8f wmma16(v16h a, v16h b, v8f c) {
  return __builtin_amdgcn_wmma_f32_16x16x32_f16(false, a, false, b, (short)0, c, false, false);
}

__device__ __forceinline__ float sigmoidf_(float x) { return 1.0f / (1.0f + __expf(-x)); }

// Branch-free tanh on v_exp_f32: exact at +/-inf, ~1e-6 rel err in active range.
__device__ __forceinline__ float fast_tanhf(float x) {
  float e = __expf(2.0f * x);
  return 1.0f - 2.0f / (e + 1.0f);
}

// ---------------------------------------------------------------------------
// Packing kernels
// ---------------------------------------------------------------------------
__global__ void cvt_f16_kernel(const float* __restrict__ src, _Float16* __restrict__ dst, int n) {
  int i = (blockIdx.x * blockDim.x + threadIdx.x) * 4;
  if (i + 3 < n) {
    float4 v = *(const float4*)(src + i);
    v4h o = { (_Float16)v.x, (_Float16)v.y, (_Float16)v.z, (_Float16)v.w };
    *(v4h*)(dst + i) = o;
  } else {
    for (; i < n; ++i) dst[i] = (_Float16)src[i];
  }
}

// f32 (rows x cols) -> f16 (rows_pad x cols), zero-filled pad rows
__global__ void cvt_pad_kernel(const float* __restrict__ src, _Float16* __restrict__ dst,
                               int rows, int rows_pad, int cols) {
  int i = blockIdx.x * blockDim.x + threadIdx.x;
  if (i >= rows_pad * cols) return;
  int r = i / cols;
  dst[i] = (r < rows) ? (_Float16)src[i] : (_Float16)0.0f;
}

// zero-padded f32 vector copy
__global__ void pad_vec_kernel(const float* __restrict__ src, float* __restrict__ dst,
                               int n, int npad) {
  int i = blockIdx.x * blockDim.x + threadIdx.x;
  if (i < npad) dst[i] = (i < n) ? src[i] : 0.0f;
}

// pad a2a_b / d2d_w (4 combos, A_ -> AP_, zeros in pad => padded cols contribute 0)
__global__ void pad_scorevec_kernel(const float* __restrict__ a2ab, const float* __restrict__ d2dw,
                                    float* __restrict__ a2ab_p, float* __restrict__ d2dw_p) {
  int i = blockIdx.x * blockDim.x + threadIdx.x;
  if (i >= 4 * AP_) return;
  int c = i / AP_, k = i % AP_;
  a2ab_p[i] = (k < A_) ? a2ab[c * A_ + k] : 0.0f;
  d2dw_p[i] = (k < A_) ? d2dw[c * A_ + k] : 0.0f;
}

__global__ void biassum_kernel(const float* __restrict__ b0, const float* __restrict__ b1,
                               const float* __restrict__ b2, float* __restrict__ out, int n) {
  int i = blockIdx.x * blockDim.x + threadIdx.x;
  if (i < n) out[i] = b0[i] + b1[i] + b2[i];
}

// ---------------------------------------------------------------------------
// hh[b,a] = h[b,:] . h2a_w[a,:] + h2a_b[a]   (small fp32 GEMV-batch)
// ---------------------------------------------------------------------------
__global__ void hh_kernel(const float* __restrict__ h, const float* __restrict__ w,
                          const float* __restrict__ bias, float* __restrict__ hh) {
  int idx = blockIdx.x * blockDim.x + threadIdx.x;  // 0..BA_-1
  int b = idx / A_, a = idx % A_;
  const float4* hp = (const float4*)(h + (size_t)b * R_);
  const float4* wp = (const float4*)(w + (size_t)a * R_);
  float acc = 0.0f;
#pragma unroll 4
  for (int i = 0; i < R_ / 4; ++i) {
    float4 xv = hp[i], yv = wp[i];
    acc += xv.x * yv.x + xv.y * yv.y + xv.z * yv.z + xv.w * yv.w;
  }
  hh[idx] = acc + bias[a];
}

// ---------------------------------------------------------------------------
// Fused score kernel: score[b,a] = sum_k tanh(att[b,a,:].a2a_w[k,:] + a2a_b[k]
//                                             + hh[b,a]) * d2d_w[k]
// One wave per 16-row tile. 13 N-tiles = 3 branch-free groups of 4 (constant
// trip counts, inner j fully unrolled) + 1 straight-line tail tile. acc[4] =
// 32 VGPRs keeps the kernel spill-free; each A-frag feeds 4 WMMAs.
// ---------------------------------------------------------------------------
__global__ void __launch_bounds__(128, 1)
score_kernel(const _Float16* __restrict__ att_h,  // (BA_, R_)
             const _Float16* __restrict__ w_h,    // (AP_, R_)
             const float* __restrict__ a2ab,      // AP_
             const float* __restrict__ d2dw,      // AP_
             const float* __restrict__ hh,        // BA_
             float* __restrict__ score) {         // BA_
  int lane = threadIdx.x & 31;
  int wave = threadIdx.x >> 5;
  int mt = blockIdx.x * (blockDim.x >> 5) + wave;
  int m0 = mt * 16;
  int col  = lane & 15;
  int rowL = (lane < 16) ? 0 : 8;

  float hhv[8];
#pragma unroll
  for (int v = 0; v < 8; ++v) hhv[v] = hh[m0 + rowL + v];

  float sacc[8];
#pragma unroll
  for (int v = 0; v < 8; ++v) sacc[v] = 0.0f;

  // 3 full groups of 4 N-tiles (branch-free)
  for (int g = 0; g < 3; ++g) {
    const _Float16* wg = w_h + (size_t)g * 64 * R_;
    v8f acc[4];
    v8f z = {};
#pragma unroll
    for (int j = 0; j < 4; ++j) acc[j] = z;

    for (int k0 = 0; k0 < R_; k0 += 32) {
      v16h a = load_frag(att_h, R_, m0, k0, lane);      // reused 4x
#pragma unroll
      for (int j = 0; j < 4; ++j) {
        v16h b = load_frag(wg, R_, j * 16, k0, lane);
        acc[j] = wmma16(a, b, acc[j]);
      }
    }
#pragma unroll
    for (int j = 0; j < 4; ++j) {
      int n = (g * 4 + j) * 16 + col;
      float bb = a2ab[n];
      float dw = d2dw[n];
#pragma unroll
      for (int v = 0; v < 8; ++v)
        sacc[v] += fast_tanhf(acc[j][v] + bb + hhv[v]) * dw;
    }
  }
  // tail tile nt = 12 (straight-line)
  {
    v8f acc = {};
    for (int k0 = 0; k0 < R_; k0 += 32) {
      v16h a = load_frag(att_h, R_, m0, k0, lane);
      v16h b = load_frag(w_h, R_, 12 * 16, k0, lane);
      acc = wmma16(a, b, acc);
    }
    float bb = a2ab[12 * 16 + col];
    float dw = d2dw[12 * 16 + col];
#pragma unroll
    for (int v = 0; v < 8; ++v)
      sacc[v] += fast_tanhf(acc[v] + bb + hhv[v]) * dw;
  }
  // Reduce the 16 column-lanes of each half-wave (rows 0..7 / 8..15)
#pragma unroll
  for (int v = 0; v < 8; ++v) {
    float s = sacc[v];
    s += __shfl_xor(s, 1, 16);
    s += __shfl_xor(s, 2, 16);
    s += __shfl_xor(s, 4, 16);
    s += __shfl_xor(s, 8, 16);
    sacc[v] = s;
  }
  if (col == 0) {
    int rbase = m0 + rowL;
#pragma unroll
    for (int v = 0; v < 8; ++v) score[rbase + v] = sacc[v];
  }
}

// ---------------------------------------------------------------------------
// Softmax over A + weighted sum of att rows. d2d_b omitted (shift-invariant).
// ---------------------------------------------------------------------------
__global__ void attres_kernel(const float* __restrict__ score, const _Float16* __restrict__ att_h,
                              float* __restrict__ out_f, _Float16* __restrict__ out_h) {
  int b = blockIdx.x, tid = threadIdx.x;
  int lane = tid & 31, wv = tid >> 5;
  __shared__ float sw[A_];
  __shared__ float red[8];

  float s = (tid < A_) ? score[b * A_ + tid] : -3.0e38f;
  float m = s;
#pragma unroll
  for (int k = 16; k >= 1; k >>= 1) m = fmaxf(m, __shfl_xor(m, k, 32));
  if (lane == 0) red[wv] = m;
  __syncthreads();
  float mx = red[0];
#pragma unroll
  for (int w = 1; w < 8; ++w) mx = fmaxf(mx, red[w]);
  __syncthreads();

  float e = (tid < A_) ? __expf(s - mx) : 0.0f;
  float es = e;
#pragma unroll
  for (int k = 16; k >= 1; k >>= 1) es += __shfl_xor(es, k, 32);
  if (lane == 0) red[wv] = es;
  __syncthreads();
  float tot = 0.0f;
#pragma unroll
  for (int w = 0; w < 8; ++w) tot += red[w];
  if (tid < A_) sw[tid] = e / tot;
  __syncthreads();

  const _Float16* base = att_h + (size_t)b * A_ * R_;
  float acc0 = 0.0f, acc1 = 0.0f;
  for (int a = 0; a < A_; ++a) {
    float wa = sw[a];
    acc0 += wa * (float)base[(size_t)a * R_ + tid];
    acc1 += wa * (float)base[(size_t)a * R_ + 256 + tid];
  }
  size_t o = (size_t)b * R_;
  out_f[o + tid] = acc0;       out_f[o + 256 + tid] = acc1;
  out_h[o + tid] = (_Float16)acc0; out_h[o + 256 + tid] = (_Float16)acc1;
}

// ---------------------------------------------------------------------------
// Gate kernel: sums[p,b,g] = xt.i2h^T + prev_h.h2h^T + att0.a2h^T + bias_sum
// Register-blocked 4 N-tiles/wave: the 3 A-frags per K-step feed 12 WMMAs.
// ---------------------------------------------------------------------------
__global__ void __launch_bounds__(128, 1)
gates_kernel(const _Float16* __restrict__ xt, const _Float16* __restrict__ ph,
             const _Float16* __restrict__ ar,
             const _Float16* __restrict__ wi, const _Float16* __restrict__ wh,
             const _Float16* __restrict__ wa,
             const float* __restrict__ bsum, float* __restrict__ sums) {
  int lane = threadIdx.x & 31;
  int wave = threadIdx.x >> 5;
  int tileId = blockIdx.x * (blockDim.x >> 5) + wave;  // 0..16*96-1
  int mt = tileId & 15, ng = tileId >> 4;              // ng: group of 4 N-tiles
  int m0 = mt * 16, n0 = ng * 64;

  v8f acc[4];
  v8f z = {};
#pragma unroll
  for (int j = 0; j < 4; ++j) acc[j] = z;

  for (int k0 = 0; k0 < R_; k0 += 32) {
    if (k0 + 32 < R_) __builtin_prefetch(wi + (size_t)n0 * R_ + k0 + 32, 0, 1);
    v16h a0 = load_frag(xt, R_, m0, k0, lane);
    v16h a1 = load_frag(ph, R_, m0, k0, lane);
    v16h a2 = load_frag(ar, R_, m0, k0, lane);
#pragma unroll
    for (int j = 0; j < 4; ++j) {
      int nb = n0 + j * 16;
      acc[j] = wmma16(a0, load_frag(wi, R_, nb, k0, lane), acc[j]);
      acc[j] = wmma16(a1, load_frag(wh, R_, nb, k0, lane), acc[j]);
      acc[j] = wmma16(a2, load_frag(wa, R_, nb, k0, lane), acc[j]);
    }
  }
  int col = lane & 15, rowL = (lane < 16) ? 0 : 8;
#pragma unroll
  for (int j = 0; j < 4; ++j) {
    int n = n0 + j * 16 + col;
    float bias = bsum[n];
    int p = n >> 11, g = n & (G4_ - 1);
#pragma unroll
    for (int v = 0; v < 8; ++v) {
      int b = m0 + rowL + v;
      sums[((size_t)(p * B_ + b)) * G4_ + g] = acc[j][v] + bias;
    }
  }
}

// ---------------------------------------------------------------------------
// LSTM cell: gates -> (next_c, next_h), mean over P banks
// ---------------------------------------------------------------------------
__global__ void cell_kernel(const float* __restrict__ sums, const float* __restrict__ prev_c,
                            float* __restrict__ next_c, float* __restrict__ next_h) {
  int idx = blockIdx.x * blockDim.x + threadIdx.x;  // 0..BR_-1
  int b = idx >> 9, r = idx & (R_ - 1);
  float pc = prev_c[idx];
  float cacc = 0.0f, hacc = 0.0f;
#pragma unroll
  for (int p = 0; p < P_; ++p) {
    const float* sp = sums + ((size_t)(p * B_ + b)) * G4_;
    float ig = sigmoidf_(sp[r]);
    float fg = sigmoidf_(sp[R_ + r]);
    float og = sigmoidf_(sp[2 * R_ + r]);
    float it = fast_tanhf(sp[3 * R_ + r]);
    float nc = fg * pc + ig * it;
    cacc += nc;
    hacc += og * fast_tanhf(nc);
  }
  next_c[idx] = cacc * (1.0f / 3.0f);
  next_h[idx] = hacc * (1.0f / 3.0f);
}

// top_h = att_res1 + next_h ; optionally xt_next = f16(x + top_h)
__global__ void toph_kernel(const float* __restrict__ ar1, const float* __restrict__ nh,
                            const float* __restrict__ x, float* __restrict__ toph,
                            _Float16* __restrict__ xt_next, _Float16* __restrict__ th_h) {
  int idx = blockIdx.x * blockDim.x + threadIdx.x;
  float t = ar1[idx] + nh[idx];
  toph[idx] = t;
  th_h[idx] = (_Float16)t;
  if (xt_next) xt_next[idx] = (_Float16)(x[idx] + t);
}

// ---------------------------------------------------------------------------
// Projection GEMM: logits[b,n] = top_h . proj_w[n,:] + proj_b_pad[n]
// V padded to 9536 = 64*149: every wave does exactly 4 N-tiles, no guards.
// ---------------------------------------------------------------------------
__global__ void __launch_bounds__(128, 1)
logits_kernel(const _Float16* __restrict__ th, const _Float16* __restrict__ pw,
              const float* __restrict__ pbp, float* __restrict__ logits) {
  int lane = threadIdx.x & 31;
  int wave = threadIdx.x >> 5;
  int tileId = blockIdx.x * (blockDim.x >> 5) + wave;  // 0..16*VGRP_-1
  int mt = tileId & 15, ng = tileId >> 4;
  int m0 = mt * 16, n0 = ng * 64;

  v8f acc[4];
  v8f z = {};
#pragma unroll
  for (int j = 0; j < 4; ++j) acc[j] = z;

  for (int k0 = 0; k0 < R_; k0 += 32) {
    v16h a = load_frag(th, R_, m0, k0, lane);
#pragma unroll
    for (int j = 0; j < 4; ++j) {
      v16h b = load_frag(pw, R_, n0 + j * 16, k0, lane);
      acc[j] = wmma16(a, b, acc[j]);
    }
  }
  int col = lane & 15, rowL = (lane < 16) ? 0 : 8;
#pragma unroll
  for (int j = 0; j < 4; ++j) {
    int n = n0 + j * 16 + col;
    float bias = pbp[n];
#pragma unroll
    for (int v = 0; v < 8; ++v) {
      int b = m0 + rowL + v;
      logits[(size_t)b * VP_ + n] = acc[j][v] + bias;
    }
  }
}

// log_softmax over V_ real columns; one block per row
__global__ void lsm_kernel(const float* __restrict__ logits, float* __restrict__ out) {
  int b = blockIdx.x, tid = threadIdx.x;
  int lane = tid & 31, wv = tid >> 5;
  __shared__ float red[8];
  const float* row = logits + (size_t)b * VP_;

  float m = -3.0e38f;
  for (int j = tid; j < V_; j += 256) m = fmaxf(m, row[j]);
#pragma unroll
  for (int k = 16; k >= 1; k >>= 1) m = fmaxf(m, __shfl_xor(m, k, 32));
  if (lane == 0) red[wv] = m;
  __syncthreads();
  float mx = red[0];
#pragma unroll
  for (int w = 1; w < 8; ++w) mx = fmaxf(mx, red[w]);
  __syncthreads();

  float s = 0.0f;
  for (int j = tid; j < V_; j += 256) s += __expf(row[j] - mx);
#pragma unroll
  for (int k = 16; k >= 1; k >>= 1) s += __shfl_xor(s, k, 32);
  if (lane == 0) red[wv] = s;
  __syncthreads();
  float tot = 0.0f;
#pragma unroll
  for (int w = 0; w < 8; ++w) tot += red[w];
  float lse = mx + __logf(tot);

  float* orow = out + (size_t)b * V_;
  for (int j = tid; j < V_; j += 256) orow[j] = row[j] - lse;
}

// ---------------------------------------------------------------------------
// Host orchestration
// ---------------------------------------------------------------------------
extern "C" void kernel_launch(void* const* d_in, const int* in_sizes, int n_in,
                              void* d_out, int out_size, void* d_ws, size_t ws_size,
                              hipStream_t stream) {
  const float* x_in     = (const float*)d_in[0];
  const float* att_in   = (const float*)d_in[1];
  const float* inputs_in= (const float*)d_in[2];
  const float* i2hw_in  = (const float*)d_in[3];
  const float* i2hb_in  = (const float*)d_in[4];
  const float* h2hw_in  = (const float*)d_in[5];
  const float* h2hb_in  = (const float*)d_in[6];
  const float* a2hw_in  = (const float*)d_in[7];
  const float* a2hb_in  = (const float*)d_in[8];
  const float* a2aw_in  = (const float*)d_in[9];
  const float* a2ab_in  = (const float*)d_in[10];
  const float* h2aw_in  = (const float*)d_in[11];
  const float* h2ab_in  = (const float*)d_in[12];
  const float* d2dw_in  = (const float*)d_in[13];
  // d_in[14] = d2d_b: scalar shift per combo -> cancels in softmax, unused
  const float* projw_in = (const float*)d_in[15];
  const float* projb_in = (const float*)d_in[16];
  float* out = (float*)d_out;

  char* ws = (char*)d_ws;
  size_t off = 0;
  auto alloc = [&](size_t bytes) -> char* {
    char* p = ws + off;
    off += (bytes + 255) & ~(size_t)255;
    return p;
  };

  _Float16* att_h  = (_Float16*)alloc((size_t)BA_ * R_ * 2);       // 51.4 MB
  _Float16* i2h_h  = (_Float16*)alloc((size_t)L_ * NG_ * R_ * 2);  // 12.6 MB
  _Float16* h2h_h  = (_Float16*)alloc((size_t)L_ * NG_ * R_ * 2);
  _Float16* a2h_h  = (_Float16*)alloc((size_t)L_ * NG_ * R_ * 2);
  _Float16* a2a_h  = (_Float16*)alloc((size_t)4 * AP_ * R_ * 2);
  _Float16* proj_h = (_Float16*)alloc((size_t)VP_ * R_ * 2);       // 9.8 MB
  float* a2ab_p = (float*)alloc(4 * AP_ * 4);
  float* d2dw_p = (float*)alloc(4 * AP_ * 4);
  float* projb_p= (float*)alloc((size_t)VP_ * 4);
  float* bsum   = (float*)alloc((size_t)L_ * NG_ * 4);
  _Float16* x_h   = (_Float16*)alloc((size_t)BR_ * 2);
  _Float16* ph_h  = (_Float16*)alloc((size_t)L_ * BR_ * 2);
  _Float16* ar0_h = (_Float16*)alloc((size_t)L_ * BR_ * 2);
  float*    ar0_f = (float*)alloc((size_t)L_ * BR_ * 4);
  float*    ar1_f = (float*)alloc((size_t)BR_ * 4);
  _Float16* ar1_h = (_Float16*)alloc((size_t)BR_ * 2);
  float* hhbuf    = (float*)alloc((size_t)BA_ * 4);
  float* scorebuf = (float*)alloc((size_t)BA_ * 4);
  float* sums     = (float*)alloc((size_t)P_ * B_ * G4_ * 4);      // 6.3 MB
  float* next_h   = (float*)alloc((size_t)BR_ * 4);
  _Float16* xt1_h = (_Float16*)alloc((size_t)BR_ * 2);
  _Float16* th_h  = (_Float16*)alloc((size_t)BR_ * 2);
  float* logits   = (float*)alloc((size_t)B_ * VP_ * 4);           // 9.8 MB
  (void)ws_size; (void)in_sizes; (void)n_in; (void)out_size;

  const int T = 256;
  const int TW = 128;  // WMMA kernels: 4 waves/block, __launch_bounds__(128,1)
  // ---- phase 0: pack to f16 ----
  cvt_f16_kernel<<<(BA_ * R_ / 4 + T - 1) / T, T, 0, stream>>>(att_in, att_h, BA_ * R_);
  cvt_f16_kernel<<<(L_ * NG_ * R_ / 4 + T - 1) / T, T, 0, stream>>>(i2hw_in, i2h_h, L_ * NG_ * R_);
  cvt_f16_kernel<<<(L_ * NG_ * R_ / 4 + T - 1) / T, T, 0, stream>>>(h2hw_in, h2h_h, L_ * NG_ * R_);
  cvt_f16_kernel<<<(L_ * NG_ * R_ / 4 + T - 1) / T, T, 0, stream>>>(a2hw_in, a2h_h, L_ * NG_ * R_);
  for (int c = 0; c < 4; ++c)
    cvt_pad_kernel<<<(AP_ * R_ + T - 1) / T, T, 0, stream>>>(
        a2aw_in + (size_t)c * A_ * R_, a2a_h + (size_t)c * AP_ * R_, A_, AP_, R_);
  cvt_pad_kernel<<<(VP_ * R_ + T - 1) / T, T, 0, stream>>>(projw_in, proj_h, V_, VP_, R_);
  pad_vec_kernel<<<(VP_ + T - 1) / T, T, 0, stream>>>(projb_in, projb_p, V_, VP_);
  pad_scorevec_kernel<<<4, T, 0, stream>>>(a2ab_in, d2dw_in, a2ab_p, d2dw_p);
  biassum_kernel<<<(L_ * NG_ + T - 1) / T, T, 0, stream>>>(i2hb_in, h2hb_in, a2hb_in, bsum, L_ * NG_);
  cvt_f16_kernel<<<(BR_ / 4 + T - 1) / T, T, 0, stream>>>(x_in, x_h, BR_);
  cvt_f16_kernel<<<(BR_ / 4 + T - 1) / T, T, 0, stream>>>(inputs_in + 1 * BR_, ph_h, BR_);
  cvt_f16_kernel<<<(BR_ / 4 + T - 1) / T, T, 0, stream>>>(inputs_in + 3 * BR_, ph_h + BR_, BR_);

  // ---- phase 1: stage-0 attention for both layers (state-independent) ----
  for (int l = 0; l < L_; ++l) {
    int c = l * 2 + 0;
    const float* h = inputs_in + (size_t)(2 * l + 1) * BR_;
    hh_kernel<<<BA_ / T, T, 0, stream>>>(h, h2aw_in + (size_t)c * A_ * R_, h2ab_in + c * A_, hhbuf);
    score_kernel<<<(BA_ / 16) / 4, TW, 0, stream>>>(att_h, a2a_h + (size_t)c * AP_ * R_,
                                                    a2ab_p + c * AP_, d2dw_p + c * AP_, hhbuf, scorebuf);
    attres_kernel<<<B_, T, 0, stream>>>(scorebuf, att_h, ar0_f + (size_t)l * BR_, ar0_h + (size_t)l * BR_);
  }

  // ---- phase 2/3: the recurrent layer chain ----
  const _Float16* xt_cur = x_h;
  for (int l = 0; l < L_; ++l) {
    gates_kernel<<<(16 * (NG_ / 64)) / 4, TW, 0, stream>>>(
        xt_cur, ph_h + (size_t)l * BR_, ar0_h + (size_t)l * BR_,
        i2h_h + (size_t)l * NG_ * R_, h2h_h + (size_t)l * NG_ * R_, a2h_h + (size_t)l * NG_ * R_,
        bsum + (size_t)l * NG_, sums);
    cell_kernel<<<BR_ / T, T, 0, stream>>>(sums, inputs_in + (size_t)(2 * l) * BR_,
                                           out + (size_t)(2 * l) * BR_, next_h);
    int c = l * 2 + 1;
    hh_kernel<<<BA_ / T, T, 0, stream>>>(next_h, h2aw_in + (size_t)c * A_ * R_, h2ab_in + c * A_, hhbuf);
    score_kernel<<<(BA_ / 16) / 4, TW, 0, stream>>>(att_h, a2a_h + (size_t)c * AP_ * R_,
                                                    a2ab_p + c * AP_, d2dw_p + c * AP_, hhbuf, scorebuf);
    attres_kernel<<<B_, T, 0, stream>>>(scorebuf, att_h, ar1_f, ar1_h);
    toph_kernel<<<BR_ / T, T, 0, stream>>>(ar1_f, next_h, x_in, out + (size_t)(2 * l + 1) * BR_,
                                           (l == 0) ? xt1_h : (_Float16*)nullptr, th_h);
    xt_cur = xt1_h;
  }

  // ---- phase 4: projection + log_softmax ----
  logits_kernel<<<(16 * VGRP_) / 4, TW, 0, stream>>>(th_h, proj_h, projb_p, logits);
  lsm_kernel<<<B_, T, 0, stream>>>(logits, out + (size_t)4 * BR_);
}
